// Interaction_12008728560154
// MI455X (gfx1250) — matrix operand order repaired and checked by reading, not compiled, not run on hardware
//
#include <hip/hip_runtime.h>
#include <hip/hip_fp16.h>

typedef __attribute__((ext_vector_type(16))) _Float16 v16h;
typedef __attribute__((ext_vector_type(8)))  _Float16 v8h;
typedef __attribute__((ext_vector_type(8)))  float    v8f;

__device__ __forceinline__ float silu_f(float x) { return x / (1.0f + __expf(-x)); }

__device__ __forceinline__ v8f vzero8() {
  v8f z;
#pragma unroll
  for (int j = 0; j < 8; ++j) z[j] = 0.0f;
  return z;
}

__device__ __forceinline__ v8f wmma16(v16h a, v16h b, v8f c) {
  // D = A(16x32 f16) * B(32x16 f16) + C(16x16 f32)
  return __builtin_amdgcn_wmma_f32_16x16x32_f16(false, a, false, b, (short)0, c, false, false);
}

// A-fragment per ISA layout: lane L -> row M=L&15; K = kb..kb+7 and kb+16..kb+23, kb=8*(L>=16).
// p points at &row[kb] in LDS; both 8-half chunks are contiguous 16B loads.
__device__ __forceinline__ v16h afrag(const _Float16* p) {
  v8h lo = *(const v8h*)(p);
  v8h hi = *(const v8h*)(p + 16);
  v16h a;
#pragma unroll
  for (int j = 0; j < 8; ++j) { a[j] = lo[j]; a[j + 8] = hi[j]; }
  return a;
}

// 3x3 helpers; compact irrep c[10] = {i, a_xy, a_xz, a_yz, s00, s11, s22, s01, s02, s12}
__device__ __forceinline__ void build3(float* T, const float* c) {
  T[0] =  c[0] + c[4]; T[1] =  c[1] + c[7]; T[2] =  c[2] + c[8];
  T[3] = -c[1] + c[7]; T[4] =  c[0] + c[5]; T[5] =  c[3] + c[9];
  T[6] = -c[2] + c[8]; T[7] = -c[3] + c[9]; T[8] =  c[0] + c[6];
}
__device__ __forceinline__ void mm3(const float* A, const float* B, float* C) {
#pragma unroll
  for (int i = 0; i < 3; ++i)
#pragma unroll
    for (int j = 0; j < 3; ++j)
      C[3 * i + j] = A[3 * i] * B[j] + A[3 * i + 1] * B[3 + j] + A[3 * i + 2] * B[6 + j];
}

// ---------------- Kernel 1: per-node attributes + channel-mixed irreps ----------------
// wave per node, lane = channel. Outputs: att_h [N][32] f16, hats [N][10][32] f32 (SoA).
__global__ __launch_bounds__(128) void k_node_prep(
    const float* __restrict__ X, const float* __restrict__ lin_w, const float* __restrict__ lin_b,
    const float* __restrict__ ln_w, const float* __restrict__ ln_b, const float* __restrict__ Wt,
    _Float16* __restrict__ att_h, float* __restrict__ hats, int N)
{
  __shared__ float s_linw[96 * 32];      // transposed: [k][h]
  __shared__ float s_wt[3 * 32 * 32];    // transposed: [comp][h][g]
  __shared__ float s_lnw[96], s_lnb[96], s_linb[32];
  __shared__ float s_x[4][96];
  __shared__ float s_c[4][32][10];

  for (int i = threadIdx.x; i < 96 * 32; i += 128) { int k = i >> 5, h = i & 31; s_linw[i] = lin_w[h * 96 + k]; }
  for (int i = threadIdx.x; i < 3 * 1024; i += 128) {
    int comp = i >> 10, h = (i >> 5) & 31, g = i & 31;
    s_wt[i] = Wt[comp * 1024 + g * 32 + h];
  }
  for (int i = threadIdx.x; i < 96; i += 128) { s_lnw[i] = ln_w[i]; s_lnb[i] = ln_b[i]; }
  if (threadIdx.x < 32) s_linb[threadIdx.x] = lin_b[threadIdx.x];
  __syncthreads();

  const int wave = threadIdx.x >> 5;
  const int lane = threadIdx.x & 31;

  for (int base = blockIdx.x * 4; base < N; base += gridDim.x * 4) {
    const int n = base + wave;
    float tI = 0.f, tA = 0.f, tS = 0.f, tX = 0.f;
    float tr3 = 0.f, a0 = 0.f, a1 = 0.f, a2 = 0.f;
    float s00 = 0.f, s11 = 0.f, s22 = 0.f, s01 = 0.f, s02 = 0.f, s12 = 0.f;
    if (n < N) {
      float t[9];
      const float* p = X + (size_t)n * 288 + lane * 9;
#pragma unroll
      for (int j = 0; j < 9; ++j) t[j] = p[j];
      tr3 = (t[0] + t[4] + t[8]) * (1.0f / 3.0f);
      a0 = 0.5f * (t[1] - t[3]); a1 = 0.5f * (t[2] - t[6]); a2 = 0.5f * (t[5] - t[7]);
      s00 = t[0] - tr3; s11 = t[4] - tr3; s22 = t[8] - tr3;
      s01 = 0.5f * (t[1] + t[3]); s02 = 0.5f * (t[2] + t[6]); s12 = 0.5f * (t[5] + t[7]);
      tI = 3.0f * tr3 * tr3;
      tA = 2.0f * (a0 * a0 + a1 * a1 + a2 * a2);
      tS = s00 * s00 + s11 * s11 + s22 * s22 + 2.0f * (s01 * s01 + s02 * s02 + s12 * s12);
#pragma unroll
      for (int j = 0; j < 9; ++j) tX += t[j] * t[j];
    }
    // LayerNorm stats over the 96 values held 3-per-lane
    float sum = tI + tA + tS;
    float sq = tI * tI + tA * tA + tS * tS;
#pragma unroll
    for (int off = 16; off > 0; off >>= 1) {
      sum += __shfl_xor(sum, off, 32);
      sq  += __shfl_xor(sq,  off, 32);
    }
    const float mu = sum * (1.0f / 96.0f);
    const float var = sq * (1.0f / 96.0f) - mu * mu;
    const float rinv = rsqrtf(var + 1e-5f);
    if (n < N) {
      s_x[wave][lane]      = (tI - mu) * rinv * s_lnw[lane]      + s_lnb[lane];
      s_x[wave][32 + lane] = (tA - mu) * rinv * s_lnw[32 + lane] + s_lnb[32 + lane];
      s_x[wave][64 + lane] = (tS - mu) * rinv * s_lnw[64 + lane] + s_lnb[64 + lane];
      const float inv = 1.0f / (tX + 1.0f);   // Xn = X / (tnorm(X)+1), decomposed compactly
      s_c[wave][lane][0] = tr3 * inv;
      s_c[wave][lane][1] = a0 * inv;  s_c[wave][lane][2] = a1 * inv;  s_c[wave][lane][3] = a2 * inv;
      s_c[wave][lane][4] = s00 * inv; s_c[wave][lane][5] = s11 * inv; s_c[wave][lane][6] = s22 * inv;
      s_c[wave][lane][7] = s01 * inv; s_c[wave][lane][8] = s02 * inv; s_c[wave][lane][9] = s12 * inv;
    }
    __syncthreads();
    if (n < N) {
      // att = silu(x @ lin_w.T + lin_b), out channel = lane
      float acc = s_linb[lane];
#pragma unroll 8
      for (int k = 0; k < 96; ++k) acc += s_linw[k * 32 + lane] * s_x[wave][k];
      att_h[(size_t)n * 32 + lane] = (_Float16)silu_f(acc);

      // channel mix of Xn irreps with Wt[0..2]; out channel g = lane
      float hi = 0.f, ha[3] = {0.f, 0.f, 0.f}, hs[6] = {0.f, 0.f, 0.f, 0.f, 0.f, 0.f};
      for (int h = 0; h < 32; ++h) {
        const float w0 = s_wt[h * 32 + lane];
        const float w1 = s_wt[1024 + h * 32 + lane];
        const float w2 = s_wt[2048 + h * 32 + lane];
        hi += w0 * s_c[wave][h][0];
#pragma unroll
        for (int k = 0; k < 3; ++k) ha[k] += w1 * s_c[wave][h][1 + k];
#pragma unroll
        for (int k = 0; k < 6; ++k) hs[k] += w2 * s_c[wave][h][4 + k];
      }
      float* hp = hats + (size_t)n * 320;
      hp[lane] = hi;
#pragma unroll
      for (int k = 0; k < 3; ++k) hp[(1 + k) * 32 + lane] = ha[k];
#pragma unroll
      for (int k = 0; k < 6; ++k) hp[(4 + k) * 32 + lane] = hs[k];
    }
    __syncthreads();
  }
}

// ---------------- Kernel 2: edge MLP 128->32->64->96 with WMMA f16 ----------------
__global__ __launch_bounds__(128) void k_edge_mlp(
    const _Float16* __restrict__ att_h, const float* __restrict__ rbf,
    const int* __restrict__ srcI, const int* __restrict__ dstI,
    const float* __restrict__ Ws0, const float* __restrict__ b0,
    const float* __restrict__ Ws1, const float* __restrict__ b1,
    const float* __restrict__ Ws2, const float* __restrict__ b2,
    float* __restrict__ eout, int E)
{
  __shared__ __align__(32) _Float16 sW0[128 * 32];   // [K][N] = Ws0^T
  __shared__ __align__(32) _Float16 sW1[32 * 64];
  __shared__ __align__(32) _Float16 sW2[64 * 96];
  __shared__ float sB0[32], sB1[64], sB2[96];
  __shared__ __align__(32) _Float16 sIn[4][16 * 128]; // per-wave input tile
  __shared__ __align__(32) _Float16 sA1[4][16 * 32];
  __shared__ __align__(32) _Float16 sA2[4][16 * 64];

  const int tid = threadIdx.x;
  for (int i = tid; i < 128 * 32; i += 128) { int k = i >> 5, n = i & 31; sW0[i] = (_Float16)Ws0[n * 128 + k]; }
  for (int i = tid; i < 32 * 64;  i += 128) { int k = i >> 6, n = i & 63; sW1[i] = (_Float16)Ws1[n * 32 + k]; }
  for (int i = tid; i < 64 * 96;  i += 128) { int k = i / 96, n = i % 96; sW2[i] = (_Float16)Ws2[n * 64 + k]; }
  for (int i = tid; i < 32; i += 128) sB0[i] = b0[i];
  for (int i = tid; i < 64; i += 128) sB1[i] = b1[i];
  for (int i = tid; i < 96; i += 128) sB2[i] = b2[i];
  __syncthreads();

  const int wave = tid >> 5;
  const int lane = tid & 31;
  const int m = lane & 15;            // A-fragment row
  const int kb = (lane >> 4) << 3;    // A-fragment K sub-offset
  const int nCol = lane & 15;         // C/D column
  const int mBase = (lane >> 4) << 3; // C/D row base

  const int numTiles = (E + 15) >> 4;
  for (int tb = blockIdx.x * 4; tb < numTiles; tb += gridDim.x * 4) {
    const int tile = tb + wave;
    const bool active = tile < numTiles;
    const int e0 = tile << 4;
    if (active) {
      // stage 16 edges x 128 f16 features: [att[src] | att[dst] | rbf]
      const int r = lane >> 1;
      const int e = e0 + r;
      _Float16* trow = &sIn[wave][r * 128];
      if (e < E) {
        if ((lane & 1) == 0) {
          const int s = srcI[e], d = dstI[e];
          const uint4* ps = (const uint4*)(att_h + (size_t)s * 32);
          const uint4* pd = (const uint4*)(att_h + (size_t)d * 32);
          uint4* t0 = (uint4*)trow;
#pragma unroll
          for (int j = 0; j < 4; ++j) t0[j] = ps[j];
#pragma unroll
          for (int j = 0; j < 4; ++j) t0[4 + j] = pd[j];
        } else {
          const float4* pr = (const float4*)(rbf + (size_t)e * 64);
          _Float16* t1 = trow + 64;
#pragma unroll
          for (int j = 0; j < 16; ++j) {
            float4 v = pr[j];
            t1[4 * j + 0] = (_Float16)v.x; t1[4 * j + 1] = (_Float16)v.y;
            t1[4 * j + 2] = (_Float16)v.z; t1[4 * j + 3] = (_Float16)v.w;
          }
        }
      }
    }
    __syncthreads();
    if (active) {
      // ---- layer 0: K=128 (4 k-steps), out 32 (2 N-tiles)
      v8f c0[2] = {vzero8(), vzero8()};
#pragma unroll
      for (int ks = 0; ks < 4; ++ks) {
        v16h a = afrag(&sIn[wave][m * 128 + ks * 32 + kb]);
#pragma unroll
        for (int t = 0; t < 2; ++t) {
          v16h b;
          for (int j = 0; j < 8; ++j) { b[j] = sW0[(ks * 32 + kb + j) * 32 + t * 16 + nCol]; b[j + 8] = sW0[(ks * 32 + kb + 16 + j) * 32 + t * 16 + nCol]; }
          c0[t] = wmma16(a, b, c0[t]);
        }
      }
#pragma unroll
      for (int t = 0; t < 2; ++t)
#pragma unroll
        for (int v = 0; v < 8; ++v) {
          float x = silu_f(c0[t][v] + sB0[t * 16 + nCol]);
          sA1[wave][(mBase + v) * 32 + t * 16 + nCol] = (_Float16)x;
        }
      // ---- layer 1: K=32, out 64 (4 N-tiles)
      v8f c1[4] = {vzero8(), vzero8(), vzero8(), vzero8()};
      {
        v16h a = afrag(&sA1[wave][m * 32 + kb]);
#pragma unroll
        for (int t = 0; t < 4; ++t) {
          v16h b;
          for (int j = 0; j < 8; ++j) { b[j] = sW1[(kb + j) * 64 + t * 16 + nCol]; b[j + 8] = sW1[(kb + 16 + j) * 64 + t * 16 + nCol]; }
          c1[t] = wmma16(a, b, c1[t]);
        }
      }
#pragma unroll
      for (int t = 0; t < 4; ++t)
#pragma unroll
        for (int v = 0; v < 8; ++v) {
          float x = silu_f(c1[t][v] + sB1[t * 16 + nCol]);
          sA2[wave][(mBase + v) * 64 + t * 16 + nCol] = (_Float16)x;
        }
      // ---- layer 2: K=64 (2 k-steps), out 96 (6 N-tiles)
      v8f c2[6] = {vzero8(), vzero8(), vzero8(), vzero8(), vzero8(), vzero8()};
#pragma unroll
      for (int ks = 0; ks < 2; ++ks) {
        v16h a = afrag(&sA2[wave][m * 64 + ks * 32 + kb]);
#pragma unroll
        for (int t = 0; t < 6; ++t) {
          v16h b;
          for (int j = 0; j < 8; ++j) { b[j] = sW2[(ks * 32 + kb + j) * 96 + t * 16 + nCol]; b[j + 8] = sW2[(ks * 32 + kb + 16 + j) * 96 + t * 16 + nCol]; }
          c2[t] = wmma16(a, b, c2[t]);
        }
      }
#pragma unroll
      for (int t = 0; t < 6; ++t)
#pragma unroll
        for (int v = 0; v < 8; ++v) {
          const int ee = e0 + mBase + v;
          if (ee < E)
            eout[(size_t)ee * 96 + t * 16 + nCol] = silu_f(c2[t][v] + sB2[t * 16 + nCol]);
        }
    }
    __syncthreads();
  }
}

// ---------------- Kernel 3: edge message scatter (compact irreps, f32 HW atomics) --------
__global__ __launch_bounds__(256) void k_scatter(
    const float* __restrict__ eout, const int* __restrict__ srcI, const int* __restrict__ dstI,
    const float* __restrict__ hats, float* __restrict__ msg, int E)
{
  const int wave = threadIdx.x >> 5;
  const int lane = threadIdx.x & 31;
  for (long long e = (long long)blockIdx.x * 8 + wave; e < (long long)E; e += (long long)gridDim.x * 8) {
    const int s = srcI[e], d = dstI[e];
    const float* hp = hats + (size_t)s * 320;
    const float* ep = eout + (size_t)e * 96 + lane * 3;
    const float f0 = ep[0], f1 = ep[1], f2 = ep[2];
    float* mp = msg + (size_t)d * 320;
    unsafeAtomicAdd(&mp[lane], f0 * hp[lane]);
#pragma unroll
    for (int k = 1; k < 4; ++k) unsafeAtomicAdd(&mp[k * 32 + lane], f1 * hp[k * 32 + lane]);
#pragma unroll
    for (int k = 4; k < 10; ++k) unsafeAtomicAdd(&mp[k * 32 + lane], f2 * hp[k * 32 + lane]);
  }
}

// ---------------- Kernel 4: O(3) combine + output ----------------
__global__ __launch_bounds__(128) void k_final(
    const float* __restrict__ X, const float* __restrict__ hats, const float* __restrict__ msg,
    const float* __restrict__ Wt, float* __restrict__ Xout, int N)
{
  __shared__ float s_wt[3 * 32 * 32];   // Wt[3..5], transposed [comp][h][g]
  __shared__ float s_c[4][32][10];
  for (int i = threadIdx.x; i < 3 * 1024; i += 128) {
    int comp = i >> 10, h = (i >> 5) & 31, g = i & 31;
    s_wt[i] = Wt[(3 + comp) * 1024 + g * 32 + h];
  }
  __syncthreads();
  const int wave = threadIdx.x >> 5;
  const int lane = threadIdx.x & 31;
  for (int base = blockIdx.x * 4; base < N; base += gridDim.x * 4) {
    const int n = base + wave;
    if (n < N) {
      const float* hp = hats + (size_t)n * 320;
      const float* mp = msg + (size_t)n * 320;
      float cy[10], cm[10];
#pragma unroll
      for (int k = 0; k < 10; ++k) { cy[k] = hp[k * 32 + lane]; cm[k] = mp[k * 32 + lane]; }
      float Y[9], M[9], B1[9], B2[9], B[9];
      build3(Y, cy); build3(M, cm);
      mm3(M, Y, B1); mm3(Y, M, B2);
      float nb = 0.f;
#pragma unroll
      for (int j = 0; j < 9; ++j) { B[j] = B1[j] + B2[j]; nb += B[j] * B[j]; }
      const float tr3 = (B[0] + B[4] + B[8]) * (1.0f / 3.0f);
      const float inv = 1.0f / (nb + 1.0f);
      s_c[wave][lane][0] = tr3 * inv;
      s_c[wave][lane][1] = 0.5f * (B[1] - B[3]) * inv;
      s_c[wave][lane][2] = 0.5f * (B[2] - B[6]) * inv;
      s_c[wave][lane][3] = 0.5f * (B[5] - B[7]) * inv;
      s_c[wave][lane][4] = (B[0] - tr3) * inv;
      s_c[wave][lane][5] = (B[4] - tr3) * inv;
      s_c[wave][lane][6] = (B[8] - tr3) * inv;
      s_c[wave][lane][7] = 0.5f * (B[1] + B[3]) * inv;
      s_c[wave][lane][8] = 0.5f * (B[2] + B[6]) * inv;
      s_c[wave][lane][9] = 0.5f * (B[5] + B[7]) * inv;
    }
    __syncthreads();
    if (n < N) {
      float dc[10];
#pragma unroll
      for (int k = 0; k < 10; ++k) dc[k] = 0.f;
      for (int h = 0; h < 32; ++h) {
        const float w3 = s_wt[h * 32 + lane];
        const float w4 = s_wt[1024 + h * 32 + lane];
        const float w5 = s_wt[2048 + h * 32 + lane];
        dc[0] += w3 * s_c[wave][h][0];
#pragma unroll
        for (int k = 1; k < 4; ++k) dc[k] += w4 * s_c[wave][h][k];
#pragma unroll
        for (int k = 4; k < 10; ++k) dc[k] += w5 * s_c[wave][h][k];
      }
      float dX[9], d2[9], t[9];
      build3(dX, dc);
      mm3(dX, dX, d2);
      const float* xp = X + (size_t)n * 288 + lane * 9;
      float tX = 0.f;
#pragma unroll
      for (int j = 0; j < 9; ++j) { t[j] = xp[j]; tX += t[j] * t[j]; }
      const float xinv = 1.0f / (tX + 1.0f);
      float* op = Xout + (size_t)n * 288 + lane * 9;
#pragma unroll
      for (int j = 0; j < 9; ++j) op[j] = t[j] * xinv + dX[j] + d2[j];
    }
    __syncthreads();
  }
}

extern "C" void kernel_launch(void* const* d_in, const int* in_sizes, int n_in,
                              void* d_out, int out_size, void* d_ws, size_t ws_size,
                              hipStream_t stream) {
  const float* X     = (const float*)d_in[0];
  const float* rbf   = (const float*)d_in[1];
  // d_in[2] = dist (unused: reference overwrites cutoff with ones)
  const int*   srcI  = (const int*)d_in[3];
  const int*   dstI  = (const int*)d_in[4];
  const float* Ws0   = (const float*)d_in[5];
  const float* b0    = (const float*)d_in[6];
  const float* Ws1   = (const float*)d_in[7];
  const float* b1    = (const float*)d_in[8];
  const float* Ws2   = (const float*)d_in[9];
  const float* b2    = (const float*)d_in[10];
  const float* Wt    = (const float*)d_in[11];
  const float* ln_w  = (const float*)d_in[12];
  const float* ln_b  = (const float*)d_in[13];
  const float* lin_w = (const float*)d_in[14];
  const float* lin_b = (const float*)d_in[15];

  const int N = in_sizes[0] / 288;   // X is [N,32,3,3]
  const int E = in_sizes[2];         // dist is [E]

  float* Xout = (float*)d_out;
  float* eout = (float*)d_out + (size_t)N * 288;

  char* ws = (char*)d_ws;
  _Float16* att_h = (_Float16*)ws;                            // N*32 f16
  size_t off = ((size_t)N * 64 + 255) & ~(size_t)255;
  float* hats = (float*)(ws + off);                           // N*320 f32
  float* msg  = hats + (size_t)N * 320;                       // N*320 f32

  const int blocksN = (N + 3) / 4;
  k_node_prep<<<blocksN, 128, 0, stream>>>(X, lin_w, lin_b, ln_w, ln_b, Wt, att_h, hats, N);

  const int numTiles = (E + 15) / 16;
  const int blocksE = (numTiles + 3) / 4;
  k_edge_mlp<<<blocksE, 128, 0, stream>>>(att_h, rbf, srcI, dstI, Ws0, b0, Ws1, b1, Ws2, b2, eout, E);

  hipMemsetAsync(msg, 0, (size_t)N * 320 * sizeof(float), stream);
  const int blocksS = (E + 7) / 8;
  k_scatter<<<blocksS, 256, 0, stream>>>(eout, srcI, dstI, hats, msg, E);

  k_final<<<blocksN, 128, 0, stream>>>(X, hats, msg, Wt, Xout, N);
}
